// MultiHeadAttention_21749714387008
// MI455X (gfx1250) — matrix operand order repaired and checked by reading, not compiled, not run on hardware
//
#include <hip/hip_runtime.h>
#include <hip/hip_bf16.h>

// ---------------------------------------------------------------------------
// MI455X (gfx1250) multi-head attention block, bf16 WMMA + fp32 accumulate.
// Shapes: S=2048 tokens, E=2048 embed, H=16 heads, D=128 head dim.
// Outputs (concatenated in d_out): out[S,E], k[S,E], v[S,E]  (all fp32).
// GEMM waves: 64x64 output tile (16 WMMA accums, ~1 b128 load per WMMA).
// Attention: one wave per (head, 32 query rows), flash-style online softmax.
// ---------------------------------------------------------------------------

#define S_TOK 2048
#define E_DIM 2048
#define H_NUM 16
#define D_HEAD 128
#define N3E (3 * E_DIM)

typedef __bf16 bf16;
typedef __attribute__((ext_vector_type(16))) __bf16 v16bf;
typedef __attribute__((ext_vector_type(8)))  float  v8f;
typedef __attribute__((ext_vector_type(4)))  unsigned int v4u;

union FragU { v4u u[2]; v16bf f; };

// A fragment (16x32 bf16, row-major source, row stride in elements).
// Lane (kh, lq): row = lq, K = kbase + kh*8 + {0..7} and kbase + 16 + kh*8 + {0..7}.
__device__ __forceinline__ v16bf load_frag_a(const bf16* __restrict__ base,
                                             int row_stride, int kbase,
                                             int lq, int kh) {
  const bf16* p = base + (size_t)lq * row_stride + kbase + kh * 8;
  FragU u;
  u.u[0] = *(const v4u*)(p);
  u.u[1] = *(const v4u*)(p + 16);
  return u.f;
}

// B fragment (32x16 bf16) from a PRE-TRANSPOSED operand Bt[N][K] (row stride
// in elements).  Lane (kh, lq): col = lq, K = kbase + kh*8 + {0..7} and kbase + 16 + kh*8 + {0..7}.
__device__ __forceinline__ v16bf load_frag_b(const bf16* __restrict__ baseT,
                                             int row_stride, int kbase,
                                             int lq, int kh) {
  const bf16* p = baseT + (size_t)lq * row_stride + kbase + kh * 8;
  FragU u;
  u.u[0] = *(const v4u*)(p);
  u.u[1] = *(const v4u*)(p + 16);
  return u.f;
}

__device__ __forceinline__ v8f wmma_bf16(v16bf a, v16bf b, v8f c) {
  return __builtin_amdgcn_wmma_f32_16x16x32_bf16(false, a, false, b,
                                                 (short)0, c, false, false);
}

// ---------------------------------------------------------------------------
// fp32 -> bf16 elementwise convert (8 elems/thread, b128 traffic)
// ---------------------------------------------------------------------------
__global__ void convert_bf16_kernel(const float* __restrict__ in,
                                    bf16* __restrict__ out, int n) {
  int i = (blockIdx.x * blockDim.x + threadIdx.x) * 8;
  if (i < n) {
#pragma unroll
    for (int j = 0; j < 8; ++j) out[i + j] = (bf16)in[i + j];
  }
}

// ---------------------------------------------------------------------------
// fp32 [K,N] -> bf16 [N,K] transposed convert (LDS tiled)
// ---------------------------------------------------------------------------
__global__ __launch_bounds__(256) void transpose_convert_kernel(
    const float* __restrict__ w, bf16* __restrict__ wt, int K, int N) {
  __shared__ float tile[32][33];
  const int tx = threadIdx.x, ty = threadIdx.y;           // block (32, 8)
  const int k0 = blockIdx.y * 32, n0 = blockIdx.x * 32;
#pragma unroll
  for (int i = 0; i < 32; i += 8)
    tile[ty + i][tx] = w[(size_t)(k0 + ty + i) * N + n0 + tx];
  __syncthreads();
#pragma unroll
  for (int i = 0; i < 32; i += 8)
    wt[(size_t)(n0 + ty + i) * K + k0 + tx] = (bf16)tile[tx][ty + i];
}

// ---------------------------------------------------------------------------
// Shared 64x64 wave-tile GEMM inner loop: acc[m][n] += A[s0+..] @ Bt[n0+..]^T
// ---------------------------------------------------------------------------
__device__ __forceinline__ void gemm_wave_64x64(const bf16* __restrict__ A,
                                                const bf16* __restrict__ Bt,
                                                int s0, int n0, int lq, int kh,
                                                v8f acc[4][4]) {
  for (int kk = 0; kk < E_DIM; kk += 32) {
    v16bf a[4];
#pragma unroll
    for (int m = 0; m < 4; ++m)
      a[m] = load_frag_a(A + (size_t)(s0 + m * 16) * E_DIM, E_DIM, kk, lq, kh);
#pragma unroll
    for (int n = 0; n < 4; ++n) {
      v16bf b = load_frag_b(Bt + (size_t)(n0 + n * 16) * E_DIM, E_DIM, kk, lq, kh);
#pragma unroll
      for (int m = 0; m < 4; ++m) acc[m][n] = wmma_bf16(a[m], b, acc[m][n]);
    }
    __builtin_prefetch(A + (size_t)(s0 + lq) * E_DIM + kk + 128, 0, 1);
  }
}

// ---------------------------------------------------------------------------
// GEMM 1: h[S,3E] = x_bf @ W_attn + b_attn, scattered into q/k/vt (+fp32 k,v).
// Block = 4 waves (2x2), block tile 128x128, wave tile 64x64.
// ---------------------------------------------------------------------------
__global__ __launch_bounds__(128) void gemm_qkv_kernel(
    const bf16* __restrict__ xb, const bf16* __restrict__ wt,
    const float* __restrict__ bias,
    bf16* __restrict__ qb, bf16* __restrict__ kb, bf16* __restrict__ vt,
    float* __restrict__ out_k, float* __restrict__ out_v) {
  const int lane = threadIdx.x & 31;
  const int wave = threadIdx.x >> 5;
  const int lq = lane & 15, kh = lane >> 4;
  const int s0 = blockIdx.y * 128 + (wave & 1) * 64;
  const int n0 = blockIdx.x * 128 + (wave >> 1) * 64;

  v8f acc[4][4] = {};
  gemm_wave_64x64(xb, wt, s0, n0, lq, kh, acc);

  const int region = n0 >> 11;  // 0=q 1=k 2=v (64-col tile stays in one region)
  const float qscale = 0.08838834764831845f;  // 1/sqrt(128)
#pragma unroll
  for (int n = 0; n < 4; ++n) {
    const int cg = n0 + n * 16 + lq;
    const int e = cg & (E_DIM - 1);
    const float bv = bias[cg];
#pragma unroll
    for (int m = 0; m < 4; ++m) {
#pragma unroll
      for (int r = 0; r < 8; ++r) {
        const int row = s0 + m * 16 + kh * 8 + r;
        const float v = acc[m][n][r] + bv;
        if (region == 0) {
          qb[(size_t)row * E_DIM + e] = (bf16)(v * qscale);
        } else if (region == 1) {
          kb[(size_t)row * E_DIM + e] = (bf16)v;
          out_k[(size_t)row * E_DIM + e] = v;
        } else {
          vt[(size_t)e * S_TOK + row] = (bf16)v;  // transposed for PV WMMA
          out_v[(size_t)row * E_DIM + e] = v;
        }
      }
    }
  }
}

// ---------------------------------------------------------------------------
// Flash attention: one wave per (head, 32 query rows).  Br=32, Bc=32.
// Per key block: 16 QK WMMAs + 16 PV WMMAs against 32 global b128 fragment loads.
// ---------------------------------------------------------------------------
__global__ __launch_bounds__(32) void attn_kernel(
    const bf16* __restrict__ qb, const bf16* __restrict__ kb,
    const bf16* __restrict__ vt, bf16* __restrict__ concat) {
  __shared__ bf16 pl[32 * 40];  // 32 rows x 32 cols, stride 40 (16B aligned)
  const int lane = threadIdx.x & 31;
  const int lq = lane & 15, kh = lane >> 4;
  const int s0 = blockIdx.x * 32;
  const int head = blockIdx.y;

  // resident q fragments: 2 M-tiles x 4 K-steps
  v16bf qf[2][4];
#pragma unroll
  for (int m = 0; m < 2; ++m) {
    const bf16* qh = qb + (size_t)(s0 + m * 16) * E_DIM + head * D_HEAD;
#pragma unroll
    for (int t = 0; t < 4; ++t) qf[m][t] = load_frag_a(qh, E_DIM, t * 32, lq, kh);
  }

  v8f o[2][8] = {};
  float mrow[2][8], lrow[2][8];
#pragma unroll
  for (int m = 0; m < 2; ++m)
#pragma unroll
    for (int r = 0; r < 8; ++r) { mrow[m][r] = -3.0e38f; lrow[m][r] = 0.0f; }

  for (int j0 = 0; j0 < s0 + 32; j0 += 32) {
    // --- scores: 2 M-tiles x 2 key tiles over keys [j0, j0+32) ---
    v8f st[2][2] = {};
#pragma unroll
    for (int t = 0; t < 4; ++t) {
#pragma unroll
      for (int nt = 0; nt < 2; ++nt) {
        v16bf b = load_frag_b(kb + (size_t)(j0 + nt * 16) * E_DIM + head * D_HEAD,
                              E_DIM, t * 32, lq, kh);
        st[0][nt] = wmma_bf16(qf[0][t], b, st[0][nt]);
        st[1][nt] = wmma_bf16(qf[1][t], b, st[1][nt]);
      }
    }
    // --- causal mask (only the diagonal block needs it) ---
    if (j0 + 31 > s0) {
#pragma unroll
      for (int m = 0; m < 2; ++m)
#pragma unroll
        for (int nt = 0; nt < 2; ++nt) {
          const int key = j0 + nt * 16 + lq;
#pragma unroll
          for (int r = 0; r < 8; ++r) {
            const int qrow = s0 + m * 16 + kh * 8 + r;
            if (key > qrow) st[m][nt][r] += -1.0e10f;
          }
        }
    }
    // --- online softmax (per-row stats live per (m, lane-half, r)) ---
#pragma unroll
    for (int m = 0; m < 2; ++m) {
#pragma unroll
      for (int r = 0; r < 8; ++r) {
        float vmax = fmaxf(st[m][0][r], st[m][1][r]);
#pragma unroll
        for (int x = 8; x >= 1; x >>= 1)
          vmax = fmaxf(vmax, __shfl_xor(vmax, x, 32));
        const float mnew = fmaxf(mrow[m][r], vmax);
        const float alpha = __expf(mrow[m][r] - mnew);
        const float p0 = __expf(st[m][0][r] - mnew);
        const float p1 = __expf(st[m][1][r] - mnew);
        st[m][0][r] = p0;
        st[m][1][r] = p1;
        float rs = p0 + p1;
#pragma unroll
        for (int x = 8; x >= 1; x >>= 1) rs += __shfl_xor(rs, x, 32);
        lrow[m][r] = lrow[m][r] * alpha + rs;
        mrow[m][r] = mnew;
#pragma unroll
        for (int n = 0; n < 8; ++n) o[m][n][r] *= alpha;
      }
    }
    // --- P: C-tile layout -> A-fragment layout via LDS ---
#pragma unroll
    for (int m = 0; m < 2; ++m)
#pragma unroll
      for (int r = 0; r < 8; ++r) {
        const int row = m * 16 + kh * 8 + r;
        pl[row * 40 + lq] = (bf16)st[m][0][r];
        pl[row * 40 + 16 + lq] = (bf16)st[m][1][r];
      }
    __syncthreads();
    v16bf pf[2];
#pragma unroll
    for (int m = 0; m < 2; ++m) {
      const bf16* p = &pl[(m * 16 + lq) * 40 + kh * 8];
      FragU u;
      u.u[0] = *(const v4u*)(p);
      u.u[1] = *(const v4u*)(p + 16);
      pf[m] = u.f;
    }
    __syncthreads();
    // --- O += P @ V  (vt is [E][S]: contiguous B fragments, reused 2x) ---
#pragma unroll
    for (int n = 0; n < 8; ++n) {
      const bf16* vp = vt + (size_t)(head * D_HEAD + n * 16 + lq) * S_TOK +
                       j0 + kh * 8;
      FragU u;
      u.u[0] = *(const v4u*)(vp);
      u.u[1] = *(const v4u*)(vp + 16);
      o[0][n] = wmma_bf16(pf[0], u.f, o[0][n]);
      o[1][n] = wmma_bf16(pf[1], u.f, o[1][n]);
    }
  }

  // --- normalize and emit concat[S,E] in bf16 for the projection GEMM ---
#pragma unroll
  for (int m = 0; m < 2; ++m)
#pragma unroll
    for (int r = 0; r < 8; ++r) {
      const float inv = 1.0f / lrow[m][r];
      const int row = s0 + m * 16 + kh * 8 + r;
#pragma unroll
      for (int n = 0; n < 8; ++n)
        concat[(size_t)row * E_DIM + head * D_HEAD + n * 16 + lq] =
            (bf16)(o[m][n][r] * inv);
    }
}

// ---------------------------------------------------------------------------
// GEMM 2: out[S,E] = concat_bf @ W_proj + b_proj  (fp32 output)
// ---------------------------------------------------------------------------
__global__ __launch_bounds__(128) void gemm_proj_kernel(
    const bf16* __restrict__ ab, const bf16* __restrict__ wt,
    const float* __restrict__ bias, float* __restrict__ out) {
  const int lane = threadIdx.x & 31;
  const int wave = threadIdx.x >> 5;
  const int lq = lane & 15, kh = lane >> 4;
  const int s0 = blockIdx.y * 128 + (wave & 1) * 64;
  const int n0 = blockIdx.x * 128 + (wave >> 1) * 64;

  v8f acc[4][4] = {};
  gemm_wave_64x64(ab, wt, s0, n0, lq, kh, acc);

#pragma unroll
  for (int n = 0; n < 4; ++n) {
    const int cg = n0 + n * 16 + lq;
    const float bv = bias[cg];
#pragma unroll
    for (int m = 0; m < 4; ++m) {
#pragma unroll
      for (int r = 0; r < 8; ++r) {
        const int row = s0 + m * 16 + kh * 8 + r;
        out[(size_t)row * E_DIM + cg] = acc[m][n][r] + bv;
      }
    }
  }
}

// ---------------------------------------------------------------------------
extern "C" void kernel_launch(void* const* d_in, const int* in_sizes, int n_in,
                              void* d_out, int out_size, void* d_ws,
                              size_t ws_size, hipStream_t stream) {
  (void)in_sizes; (void)n_in; (void)out_size; (void)ws_size;
  const float* x = (const float*)d_in[0];
  const float* W_attn = (const float*)d_in[1];
  const float* b_attn = (const float*)d_in[2];
  const float* W_proj = (const float*)d_in[3];
  const float* b_proj = (const float*)d_in[4];
  float* out = (float*)d_out;

  char* ws = (char*)d_ws;
  size_t off = 0;
  bf16* x_bf = (bf16*)(ws + off);     off += (size_t)S_TOK * E_DIM * 2;
  bf16* wattn_t = (bf16*)(ws + off);  off += (size_t)N3E * E_DIM * 2;
  bf16* wproj_t = (bf16*)(ws + off);  off += (size_t)E_DIM * E_DIM * 2;
  bf16* q_bf = (bf16*)(ws + off);     off += (size_t)S_TOK * E_DIM * 2;
  bf16* k_bf = (bf16*)(ws + off);     off += (size_t)S_TOK * E_DIM * 2;
  bf16* vt_bf = (bf16*)(ws + off);    off += (size_t)E_DIM * S_TOK * 2;
  bf16* concat_bf = (bf16*)(ws + off);

  const int nX = S_TOK * E_DIM;
  convert_bf16_kernel<<<nX / (256 * 8), 256, 0, stream>>>(x, x_bf, nX);
  transpose_convert_kernel<<<dim3(N3E / 32, E_DIM / 32), dim3(32, 8), 0, stream>>>(
      W_attn, wattn_t, E_DIM, N3E);
  transpose_convert_kernel<<<dim3(E_DIM / 32, E_DIM / 32), dim3(32, 8), 0, stream>>>(
      W_proj, wproj_t, E_DIM, E_DIM);

  gemm_qkv_kernel<<<dim3(N3E / 128, S_TOK / 128), 128, 0, stream>>>(
      x_bf, wattn_t, b_attn, q_bf, k_bf, vt_bf,
      out + (size_t)S_TOK * E_DIM, out + (size_t)2 * S_TOK * E_DIM);

  attn_kernel<<<dim3(S_TOK / 32, H_NUM), 32, 0, stream>>>(q_bf, k_bf, vt_bf,
                                                          concat_bf);

  gemm_proj_kernel<<<dim3(E_DIM / 128, S_TOK / 128), 128, 0, stream>>>(
      concat_bf, wproj_t, b_proj, out);
}
